// SelfAttention_71554155152053
// MI455X (gfx1250) — matrix operand order repaired and checked by reading, not compiled, not run on hardware
//
#include <hip/hip_runtime.h>

// ---------------------------------------------------------------------------
// FlashAttention-style fused attention, B=2,H=8,S=4096,D=64, fp32 in/out.
//
// Pass 1 (once, into d_ws): K -> f16 row-major K16[bh][s][d];
//                           V -> f16 transposed VT16[bh][d][s] (LDS transpose).
// Pass 2: one wave32 per 16-query tile, 64 keys/iteration,
//   18 v_wmma_f32_16x16x32_f16 per loop body (8 QK^T + 8 PV + 2 row-sum).
//   K/V fragments are direct v16h (2x b128) loads -- no cvt, no LDS staging.
//   Softmax row-max via DPP ROW_XMASK; row-sum via P x ones WMMA tile.
//   Scale (1/16)*log2(e) folded into Q->f16; exp2 on raw v_exp_f32.
// ---------------------------------------------------------------------------

typedef __attribute__((ext_vector_type(16))) _Float16 v16h;
typedef __attribute__((ext_vector_type(8)))  _Float16 v8h;
typedef __attribute__((ext_vector_type(8)))  float    v8f;

#define S_LEN 4096
#define D_DIM 64
#define NBH   16              // B*H
#define KT    64              // keys per main-loop iteration
#define PSTR  72              // halves; 144B rows -> 16B aligned, bank-padded
#define P_H   (16 * PSTR)     // per-wave P scratch
#define TSTR  72              // transpose-tile LDS stride (halves)

__device__ __forceinline__ float4 ld4(const float* p) {
  return *reinterpret_cast<const float4*>(p);
}

// Two runs of 8 contiguous fp32, pre-scaled, -> 16-half A-fragment register.
__device__ __forceinline__ v16h cvt8x2s(const float* p, const float* q, float sc) {
  float4 a = ld4(p), b = ld4(p + 4), c = ld4(q), d = ld4(q + 4);
  v16h r;
  r[0]  = (_Float16)(a.x*sc); r[1]  = (_Float16)(a.y*sc); r[2]  = (_Float16)(a.z*sc); r[3]  = (_Float16)(a.w*sc);
  r[4]  = (_Float16)(b.x*sc); r[5]  = (_Float16)(b.y*sc); r[6]  = (_Float16)(b.z*sc); r[7]  = (_Float16)(b.w*sc);
  r[8]  = (_Float16)(c.x*sc); r[9]  = (_Float16)(c.y*sc); r[10] = (_Float16)(c.z*sc); r[11] = (_Float16)(c.w*sc);
  r[12] = (_Float16)(d.x*sc); r[13] = (_Float16)(d.y*sc); r[14] = (_Float16)(d.z*sc); r[15] = (_Float16)(d.w*sc);
  return r;
}

// 16 contiguous fp32 -> 16 halves (fallback path only).
__device__ __forceinline__ v16h cvt16(const float* p) { return cvt8x2s(p, p + 8, 1.0f); }

__device__ __forceinline__ v16h cat8(v8h a, v8h b) {
  v16h r;
#pragma unroll
  for (int i = 0; i < 8; ++i) { r[i] = a[i]; r[i + 8] = b[i]; }
  return r;
}

// Max across the 16-lane row group using DPP ROW_XMASK (0x160|mask): pure VALU.
#define DPP_MAX_STEP(v, mask)                                                   \
  v = fmaxf(v, __int_as_float(__builtin_amdgcn_update_dpp(                      \
          0, __float_as_int(v), 0x160 | (mask), 0xf, 0xf, true)))

__device__ __forceinline__ float rowmax16(float v) {
  DPP_MAX_STEP(v, 1);
  DPP_MAX_STEP(v, 2);
  DPP_MAX_STEP(v, 4);
  DPP_MAX_STEP(v, 8);
  return v;
}

// ======================= Pass 1a: K fp32 -> f16 copy ========================
__global__ __launch_bounds__(256, 4)
void k16_convert_kernel(const float* __restrict__ K, _Float16* __restrict__ K16) {
  const size_t i = ((size_t)blockIdx.x * 256 + threadIdx.x) * 8;  // 8 elems/thread
  float4 a = ld4(K + i), b = ld4(K + i + 4);
  v8h r;
  r[0] = (_Float16)a.x; r[1] = (_Float16)a.y; r[2] = (_Float16)a.z; r[3] = (_Float16)a.w;
  r[4] = (_Float16)b.x; r[5] = (_Float16)b.y; r[6] = (_Float16)b.z; r[7] = (_Float16)b.w;
  *reinterpret_cast<v8h*>(K16 + i) = r;
}

// ================ Pass 1b: V fp32 -> f16 transposed [bh][d][s] ==============
// Block = one 64-key x 64-dim tile; LDS transpose keeps both sides coalesced.
__global__ __launch_bounds__(256, 2)
void vt_transpose_kernel(const float* __restrict__ V, _Float16* __restrict__ VT) {
  __shared__ __align__(16) _Float16 t[D_DIM * TSTR];
  const int tid = threadIdx.x;
  const int bh  = blockIdx.x >> 6;       // 0..15
  const int st  = blockIdx.x & 63;       // s-tile, 64 keys each
  // Read: 4 threads per key; each converts 16 dims, scatter halves into LDS.
  {
    const int key = tid >> 2, d0 = (tid & 3) * 16;
    const float* vrow = V + (((size_t)bh * S_LEN + st * 64 + key) * D_DIM) + d0;
#pragma unroll
    for (int i = 0; i < 16; i += 4) {
      float4 x = ld4(vrow + i);
      const int d = d0 + i;
      t[(d + 0) * TSTR + key] = (_Float16)x.x;
      t[(d + 1) * TSTR + key] = (_Float16)x.y;
      t[(d + 2) * TSTR + key] = (_Float16)x.z;
      t[(d + 3) * TSTR + key] = (_Float16)x.w;
    }
  }
  __syncthreads();
  // Write: 4 threads per dim; 32B contiguous runs along s (coalesced).
  {
    const int dim = tid >> 2, s0 = (tid & 3) * 16;
    v16h row = *reinterpret_cast<const v16h*>(&t[dim * TSTR + s0]);
    *reinterpret_cast<v16h*>(VT + ((size_t)bh * D_DIM + dim) * S_LEN + st * 64 + s0) = row;
  }
}

// ===================== Pass 2: fused attention (f16 K/V) ====================
__global__ __launch_bounds__(256, 1)
void SelfAttention_71554155152053_kernel(const float* __restrict__ Q,
                                         const _Float16* __restrict__ K16,
                                         const _Float16* __restrict__ VT16,
                                         float* __restrict__ O) {
  __shared__ __align__(16) _Float16 pl_all[8 * P_H];   // per-wave P scratch only

  const int tid  = threadIdx.x;
  const int wave = tid >> 5;
  const int lane = tid & 31;
  const int g    = lane >> 4;      // half-wave group (0/1)
  const int ln   = lane & 15;      // lane-in-group

  const int bh   = blockIdx.x >> 5;          // 0..15  (B*H)
  const int qblk = blockIdx.x & 31;          // 0..31  (128 queries per block)
  const int q0   = qblk * 128 + wave * 16;

  const float*    Qb = Q    + (size_t)bh * S_LEN * D_DIM;
  const _Float16* Kb = K16  + (size_t)bh * S_LEN * D_DIM;
  const _Float16* Vb = VT16 + (size_t)bh * D_DIM * S_LEN;   // [d][s]
  float*          Ob = O    + (size_t)bh * S_LEN * D_DIM;

  _Float16* pl = pl_all + wave * P_H;

  // ---- Q tile -> two A-fragments, pre-scaled by (1/16)*log2(e) ----
  const float qscale = 0.0625f * 1.44269504088896340736f;
  v16h qa0, qa1;
  {
    const float* qrow = Qb + (size_t)(q0 + ln) * D_DIM + g * 16;
    qa0 = cvt8x2s(qrow,      qrow + 8, qscale);
    qa1 = cvt8x2s(qrow + 32, qrow + 40, qscale);
  }

  // Constant all-ones B fragment: P x ones accumulates softmax row sums.
  v16h ones;
#pragma unroll
  for (int i = 0; i < 16; ++i) ones[i] = (_Float16)1.0f;

  v8f acc[4] = {v8f{}, v8f{}, v8f{}, v8f{}};
  v8f lacc   = {};
  float m[8];
#pragma unroll
  for (int r = 0; r < 8; ++r) m[r] = -__builtin_inff();

  for (int j0 = 0; j0 < S_LEN; j0 += KT) {
    // ---- Prefetch next K/V tiles toward L2 (global_prefetch_b8) ----
    if (j0 + KT < S_LEN) {
      __builtin_prefetch(Kb + (size_t)(j0 + KT + 2 * lane) * D_DIM, 0, 0);
      __builtin_prefetch(Vb + (size_t)(2 * lane) * S_LEN + j0 + KT, 0, 0);
    }

    // ---- Scores: S = (Q*scale*log2e) K^T, four 16-key tiles ----
    // K B-fragment = 16 contiguous halves: direct v16h load, no cvt.
    v8f s[4];
#pragma unroll
    for (int t = 0; t < 4; ++t) {
      const _Float16* krow = Kb + (size_t)(j0 + t * 16 + ln) * D_DIM + g * 16;
      v16h b0 = *reinterpret_cast<const v16h*>(krow);
      v8f z = {};
      v8f p = __builtin_amdgcn_wmma_f32_16x16x32_f16(false, qa0, false, b0,
                                                     (short)0, z, false, false);
      v16h b1 = *reinterpret_cast<const v16h*>(krow + 32);
      s[t] = __builtin_amdgcn_wmma_f32_16x16x32_f16(false, qa1, false, b1,
                                                    (short)0, p, false, false);
    }

    // ---- Online softmax (log2 domain); sums go through the matrix unit ----
    float alpha[8];
#pragma unroll
    for (int r = 0; r < 8; ++r) {
      float s0 = s[0][r], s1 = s[1][r], s2 = s[2][r], s3 = s[3][r];
      float t = fmaxf(fmaxf(s0, s1), fmaxf(s2, s3));
      t = rowmax16(t);
      float mn = fmaxf(m[r], t);
      alpha[r] = __builtin_amdgcn_exp2f(m[r] - mn);
      m[r] = mn;
      float p0 = __builtin_amdgcn_exp2f(s0 - mn);
      float p1 = __builtin_amdgcn_exp2f(s1 - mn);
      float p2 = __builtin_amdgcn_exp2f(s2 - mn);
      float p3 = __builtin_amdgcn_exp2f(s3 - mn);
      _Float16* prow = pl + (r + 8 * g) * PSTR + ln;
      prow[0]  = (_Float16)p0;
      prow[16] = (_Float16)p1;
      prow[32] = (_Float16)p2;
      prow[48] = (_Float16)p3;
    }

    // ---- Rescale accumulators (incl. row-sum tile) by alpha ----
#pragma unroll
    for (int r = 0; r < 8; ++r) {
      acc[0][r] *= alpha[r]; acc[1][r] *= alpha[r];
      acc[2][r] *= alpha[r]; acc[3][r] *= alpha[r];
      lacc[r]   *= alpha[r];
    }

    asm volatile("s_wait_dscnt 0" ::: "memory");   // wave-private P fence

    // ---- O += P*V, lsum += P*ones : depth 64 keys = 2 chunks of 32 ----
#pragma unroll
    for (int c = 0; c < 2; ++c) {
      const _Float16* prow = pl + ln * PSTR + c * 32 + g * 16;
      v16h pa = cat8(*reinterpret_cast<const v8h*>(prow),
                     *reinterpret_cast<const v8h*>(prow + 8));
      // V B-fragment = 16 contiguous halves of VT16[dim][keys]: direct load.
#pragma unroll
      for (int t = 0; t < 4; ++t) {
        const _Float16* vcol = Vb + (size_t)(t * 16 + ln) * S_LEN + j0 + c * 32 + g * 16;
        v16h bf = *reinterpret_cast<const v16h*>(vcol);
        acc[t] = __builtin_amdgcn_wmma_f32_16x16x32_f16(false, pa, false, bf,
                                                        (short)0, acc[t], false, false);
      }
      lacc = __builtin_amdgcn_wmma_f32_16x16x32_f16(false, pa, false, ones,
                                                    (short)0, lacc, false, false);
    }
  }

  float rl[8];
#pragma unroll
  for (int r = 0; r < 8; ++r) rl[r] = 1.0f / lacc[r];
#pragma unroll
  for (int t = 0; t < 4; ++t) {
#pragma unroll
    for (int r = 0; r < 8; ++r) {
      Ob[(size_t)(q0 + r + 8 * g) * D_DIM + t * 16 + ln] = acc[t][r] * rl[r];
    }
  }
}

// ============== Fallback: fp32-direct variant (ws too small) ================
__global__ __launch_bounds__(256, 1)
void SelfAttention_f32_fallback_kernel(const float* __restrict__ Q,
                                       const float* __restrict__ K,
                                       const float* __restrict__ V,
                                       float* __restrict__ O) {
#define VSTR 72
#define VT_H (D_DIM * VSTR)
  __shared__ __align__(16) _Float16 lds[2 * VT_H + 8 * P_H];
  const int tid  = threadIdx.x;
  const int wave = tid >> 5;
  const int lane = tid & 31;
  const int g    = lane >> 4;
  const int ln   = lane & 15;
  const int bh   = blockIdx.x >> 5;
  const int qblk = blockIdx.x & 31;
  const int q0   = qblk * 128 + wave * 16;
  const size_t base = (size_t)bh * S_LEN * D_DIM;
  const float* Qb = Q + base;
  const float* Kb = K + base;
  const float* Vb = V + base;
  float*       Ob = O + base;
  _Float16* pl = lds + 2 * VT_H + wave * P_H;

  const float qscale = 0.0625f * 1.44269504088896340736f;
  v16h qa0, qa1;
  {
    const float* qrow = Qb + (size_t)(q0 + ln) * D_DIM + g * 16;
    qa0 = cvt8x2s(qrow,      qrow + 8, qscale);
    qa1 = cvt8x2s(qrow + 32, qrow + 40, qscale);
  }
  v16h ones;
#pragma unroll
  for (int i = 0; i < 16; ++i) ones[i] = (_Float16)1.0f;
  v8f acc[4] = {v8f{}, v8f{}, v8f{}, v8f{}};
  v8f lacc   = {};
  float m[8];
#pragma unroll
  for (int r = 0; r < 8; ++r) m[r] = -__builtin_inff();
  const int vkey = tid >> 2, vd0 = (tid & 3) * 16;

  for (int it = 0; it < S_LEN / KT; ++it) {
    const int j0 = it * KT;
    _Float16* vt = lds + (it & 1) * VT_H;
    {
      const float* vrow = Vb + (size_t)(j0 + vkey) * D_DIM + vd0;
#pragma unroll
      for (int i = 0; i < 16; i += 4) {
        float4 x = ld4(vrow + i);
        const int d = vd0 + i;
        vt[(d + 0) * VSTR + vkey] = (_Float16)x.x;
        vt[(d + 1) * VSTR + vkey] = (_Float16)x.y;
        vt[(d + 2) * VSTR + vkey] = (_Float16)x.z;
        vt[(d + 3) * VSTR + vkey] = (_Float16)x.w;
      }
    }
    __syncthreads();
    v8f s[4];
#pragma unroll
    for (int t = 0; t < 4; ++t) {
      const float* krow = Kb + (size_t)(j0 + t * 16 + ln) * D_DIM + g * 16;
      v16h b0 = cvt16(krow);
      v8f z = {};
      v8f p = __builtin_amdgcn_wmma_f32_16x16x32_f16(false, qa0, false, b0,
                                                     (short)0, z, false, false);
      v16h b1 = cvt16(krow + 32);
      s[t] = __builtin_amdgcn_wmma_f32_16x16x32_f16(false, qa1, false, b1,
                                                    (short)0, p, false, false);
    }
    float alpha[8];
#pragma unroll
    for (int r = 0; r < 8; ++r) {
      float s0 = s[0][r], s1 = s[1][r], s2 = s[2][r], s3 = s[3][r];
      float t = fmaxf(fmaxf(s0, s1), fmaxf(s2, s3));
      t = rowmax16(t);
      float mn = fmaxf(m[r], t);
      alpha[r] = __builtin_amdgcn_exp2f(m[r] - mn);
      m[r] = mn;
      float p0 = __builtin_amdgcn_exp2f(s0 - mn);
      float p1 = __builtin_amdgcn_exp2f(s1 - mn);
      float p2 = __builtin_amdgcn_exp2f(s2 - mn);
      float p3 = __builtin_amdgcn_exp2f(s3 - mn);
      _Float16* prow = pl + (r + 8 * g) * PSTR + ln;
      prow[0]  = (_Float16)p0;
      prow[16] = (_Float16)p1;
      prow[32] = (_Float16)p2;
      prow[48] = (_Float16)p3;
    }
#pragma unroll
    for (int r = 0; r < 8; ++r) {
      acc[0][r] *= alpha[r]; acc[1][r] *= alpha[r];
      acc[2][r] *= alpha[r]; acc[3][r] *= alpha[r];
      lacc[r]   *= alpha[r];
    }
    asm volatile("s_wait_dscnt 0" ::: "memory");
#pragma unroll
    for (int c = 0; c < 2; ++c) {
      const _Float16* prow = pl + ln * PSTR + c * 32 + g * 16;
      v16h pa = cat8(*reinterpret_cast<const v8h*>(prow),
                     *reinterpret_cast<const v8h*>(prow + 8));
#pragma unroll
      for (int t = 0; t < 4; ++t) {
        const _Float16* vcol = vt + (t * 16 + ln) * VSTR + c * 32 + g * 16;
        v16h bf = cat8(*reinterpret_cast<const v8h*>(vcol),
                       *reinterpret_cast<const v8h*>(vcol + 8));
        acc[t] = __builtin_amdgcn_wmma_f32_16x16x32_f16(false, pa, false, bf,
                                                        (short)0, acc[t], false, false);
      }
      lacc = __builtin_amdgcn_wmma_f32_16x16x32_f16(false, pa, false, ones,
                                                    (short)0, lacc, false, false);
    }
  }
  float rl[8];
#pragma unroll
  for (int r = 0; r < 8; ++r) rl[r] = 1.0f / lacc[r];
#pragma unroll
  for (int t = 0; t < 4; ++t) {
#pragma unroll
    for (int r = 0; r < 8; ++r) {
      Ob[(size_t)(q0 + r + 8 * g) * D_DIM + t * 16 + ln] = acc[t][r] * rl[r];
    }
  }
#undef VSTR
#undef VT_H
}

extern "C" void kernel_launch(void* const* d_in, const int* in_sizes, int n_in,
                              void* d_out, int out_size, void* d_ws, size_t ws_size,
                              hipStream_t stream) {
  const float* q = (const float*)d_in[0];
  const float* k = (const float*)d_in[1];
  const float* v = (const float*)d_in[2];
  float* o = (float*)d_out;

  const size_t elems = (size_t)NBH * S_LEN * D_DIM;          // 4,194,304
  const size_t need  = 2 * elems * sizeof(_Float16);         // 16 MB

  if (ws_size >= need) {
    _Float16* k16 = (_Float16*)d_ws;
    _Float16* vt16 = k16 + elems;
    // Pass 1: precision conversion + V transpose (one-time, HBM-bound)
    hipLaunchKernelGGL(k16_convert_kernel, dim3((unsigned)(elems / (256 * 8))),
                       dim3(256), 0, stream, k, k16);
    hipLaunchKernelGGL(vt_transpose_kernel, dim3(NBH * (S_LEN / 64)),
                       dim3(256), 0, stream, v, vt16);
    // Pass 2: fused attention
    hipLaunchKernelGGL(SelfAttention_71554155152053_kernel, dim3(512), dim3(256),
                       0, stream, q, k16, vt16, o);
  } else {
    hipLaunchKernelGGL(SelfAttention_f32_fallback_kernel, dim3(512), dim3(256),
                       0, stream, q, k, v, o);
  }
}
